// STConvBlock_45440753992502
// MI455X (gfx1250) — matrix operand order repaired and checked by reading, not compiled, not run on hardware
//
#include <hip/hip_runtime.h>
#include <cstdint>
#include <cstddef>

// ---------------------------------------------------------------------------
// CDNA5 / gfx1250 fused STConv block.
//   stage12 kernel: temporal GLU conv #1 + full Chebyshev chain, fused per
//                   (b,t) workgroup; cur matrix kept in 132KB of LDS (bf16,
//                   transposed, padded pitch), chained with v_wmma_f32_16x16x32_bf16.
//                   Register-pressure-shaped: one 16-row strip of fp32
//                   accumulators live at a time; results parked as packed bf16.
//   stage3 kernel:  temporal GLU conv #2 as im2col bf16 WMMA + fused GLU.
// ---------------------------------------------------------------------------

typedef __bf16        v16bf __attribute__((ext_vector_type(16)));
typedef float         v8f   __attribute__((ext_vector_type(8)));
typedef unsigned int  u32x4 __attribute__((ext_vector_type(4)));

union BfTile { v16bf bf; u32x4 u[2]; };

__device__ __forceinline__ unsigned short f2bf(float f) {
  unsigned int u = __float_as_uint(f);
  u += 0x7FFFu + ((u >> 16) & 1u);          // round-to-nearest-even
  return (unsigned short)(u >> 16);
}
__device__ __forceinline__ unsigned int pack2bf(float a, float b) {
  return (unsigned int)f2bf(a) | ((unsigned int)f2bf(b) << 16);
}
__device__ __forceinline__ float bf2f(unsigned short h) {
  return __uint_as_float(((unsigned int)h) << 16);
}

#define NN          1024   // graph nodes
#define CTEMP       64     // temporal_dim
#define CSPAT       16     // spatial_dim
#define T_IN        64
#define T1          62
#define T2          60
#define KC          3
#define CURT_PITCH  1032   // 1024 + 8 pad (2064B rows: 16B-aligned, bank-staggered)
#define W2T_PITCH   72     // 64 + 8 pad

// ---------------------------------------------------------------------------
// cheb fp32 -> bf16 (one-time; 12.6MB read, 6.3MB write; lands in L2)
// ---------------------------------------------------------------------------
__global__ void cheb_to_bf16(const float* __restrict__ cheb,
                             unsigned short* __restrict__ chebb, int n) {
  int i = blockIdx.x * blockDim.x + threadIdx.x;
  if (i < n) chebb[i] = f2bf(cheb[i]);
}

// ---------------------------------------------------------------------------
// Fused stage 1 + stage 2.  One 1024-thread workgroup per (b,t).
// ---------------------------------------------------------------------------
__global__ void __launch_bounds__(1024)
stconv_stage12(const float* __restrict__ x,
               const unsigned short* __restrict__ chebb,
               const float* __restrict__ W1,
               const float* __restrict__ b1,
               const float* __restrict__ Theta,
               unsigned short* __restrict__ h2b)
{
  __shared__ unsigned short sCurT[CTEMP * CURT_PITCH];  // 132,096 B: cur^T [c][n] bf16
  __shared__ float sTheta[KC * CTEMP * CSPAT];          // 12,288 B
  __shared__ float sW1[3 * 2 * 2 * CTEMP];              // 3,072 B
  __shared__ float sB1[2 * CTEMP];                      // 512 B

  const int tid = threadIdx.x;
  const int bt  = blockIdx.x;
  const int b   = bt / T1, t = bt % T1;

  for (int i = tid; i < 3 * 2 * 2 * CTEMP;  i += 1024) sW1[i]    = W1[i];
  for (int i = tid; i < 2 * CTEMP;          i += 1024) sB1[i]    = b1[i];
  for (int i = tid; i < KC * CTEMP * CSPAT; i += 1024) sTheta[i] = Theta[i];
  __syncthreads();

  // ---- stage 1: h1[b,t,n,:] = GLU(conv) directly into LDS (thread = node n)
  {
    const int n = tid;
    float xv[3][2];
    #pragma unroll
    for (int kt = 0; kt < 3; ++kt) {
      const float* xp = x + ((size_t)(b * T_IN + t + kt) * NN + n) * 2;
      xv[kt][0] = xp[0]; xv[kt][1] = xp[1];
    }
    for (int co = 0; co < CTEMP; ++co) {
      float p = sB1[co], q = sB1[CTEMP + co];
      #pragma unroll
      for (int kt = 0; kt < 3; ++kt)
        #pragma unroll
        for (int ci = 0; ci < 2; ++ci) {
          const float w = xv[kt][ci];
          p += w * sW1[(kt * 2 + ci) * 128 + co];
          q += w * sW1[(kt * 2 + ci) * 128 + 64 + co];
        }
      const float g = p * (1.0f / (1.0f + __expf(-q)));
      sCurT[co * CURT_PITCH + n] = f2bf(g);
    }
  }
  __syncthreads();

  // ---- stage 2: k-chain.  wave owns M rows [wid*32, wid*32+32) -> 2x4 tiles.
  const int lane = tid & 31, lo = lane & 15, hi = lane >> 4;
  const int wid  = tid >> 5;
  const int mbase = wid * 32;

  float acc[CSPAT];
  #pragma unroll
  for (int j = 0; j < CSPAT; ++j) acc[j] = 0.0f;

  const v8f vz = {0.f,0.f,0.f,0.f,0.f,0.f,0.f,0.f};

  for (int k = 0; k < KC; ++k) {
    const unsigned short* chk = chebb + (size_t)k * NN * NN;

    // packed-bf16 parking for new_cur: 2 strips x 4 ct-tiles x 4 row-pairs
    unsigned int pend[2][4][4];

    // read phase: new_cur = cheb[k] @ cur.  One strip of fp32 acc at a time.
    #pragma unroll
    for (int mt = 0; mt < 2; ++mt) {
      const int m0 = mbase + mt * 16;
      // A layout (16-bit 16x32): lane holds row m0+lo, K = 8*hi+{0..7} | 16+8*hi+{0..7}
      const unsigned short* arow = chk + (size_t)(m0 + lo) * NN + 8 * hi;
      const unsigned short* bcol = &sCurT[lo * CURT_PITCH + 8 * hi];

      v8f d[4];
      #pragma unroll
      for (int ct = 0; ct < 4; ++ct) d[ct] = vz;

      #pragma unroll 2
      for (int kk = 0; kk < NN; kk += 32) {
        BfTile A;
        A.u[0] = *(const u32x4*)(arow + kk);
        A.u[1] = *(const u32x4*)(arow + kk + 16);
        #pragma unroll
        for (int ct = 0; ct < 4; ++ct) {
          // B layout (16-bit 32x16): lane holds col ct*16+lo, K = 8*hi+{0..7} | 16+8*hi+{0..7}
          BfTile B;
          const unsigned short* bp = bcol + ct * (16 * CURT_PITCH) + kk;
          B.u[0] = *(const u32x4*)(bp);
          B.u[1] = *(const u32x4*)(bp + 16);
          d[ct] = __builtin_amdgcn_wmma_f32_16x16x32_bf16(
              false, A.bf, false, B.bf, (short)0, d[ct], false, false);
        }
      }
      // park this strip as packed bf16 row-pairs (fp32 regs die here)
      #pragma unroll
      for (int ct = 0; ct < 4; ++ct)
        #pragma unroll
        for (int j = 0; j < 4; ++j)
          pend[mt][ct][j] = pack2bf(d[ct][2 * j], d[ct][2 * j + 1]);
    }
    __syncthreads();   // all reads of cur_k complete before overwriting

    // write phase: cur <- new_cur.  D VGPR v -> row (8*hi + v); pairs (2j,2j+1)
    // are adjacent rows at the same column -> aligned packed b32 stores.
    #pragma unroll
    for (int mt = 0; mt < 2; ++mt)
      #pragma unroll
      for (int ct = 0; ct < 4; ++ct)
        #pragma unroll
        for (int j = 0; j < 4; ++j) {
          const int m = mbase + mt * 16 + 8 * hi + 2 * j;
          const int c = ct * 16 + lo;
          *(unsigned int*)(&sCurT[c * CURT_PITCH + m]) = pend[mt][ct][j];
        }
    __syncthreads();

    // acc += new_cur @ Theta[k]   (tiny: 64x16 per row, VALU; thread = row n)
    const float* th = &sTheta[k * CTEMP * CSPAT];
    const int n = tid;
    for (int c = 0; c < CTEMP; ++c) {
      const float cv = bf2f(sCurT[c * CURT_PITCH + n]);
      #pragma unroll
      for (int j = 0; j < CSPAT; ++j) acc[j] += cv * th[c * CSPAT + j];
    }
  }

  // ---- ReLU + store h2 as bf16 (packed 32B per node)
  {
    const int n = tid;
    const size_t o = ((size_t)(b * T1 + t) * NN + n) * CSPAT;
    unsigned int pk[8];
    #pragma unroll
    for (int j = 0; j < 8; ++j) {
      float a0 = acc[2 * j];     a0 = a0 > 0.f ? a0 : 0.f;
      float a1 = acc[2 * j + 1]; a1 = a1 > 0.f ? a1 : 0.f;
      pk[j] = pack2bf(a0, a1);
    }
    u32x4 w0 = {pk[0], pk[1], pk[2], pk[3]};
    u32x4 w1 = {pk[4], pk[5], pk[6], pk[7]};
    *(u32x4*)(h2b + o)     = w0;
    *(u32x4*)(h2b + o + 8) = w1;
  }
}

// ---------------------------------------------------------------------------
// Stage 3: out = GLU(im2col(h2) @ W2 + b2).  M=16x60x1024 rows, K=48 (pad 64),
// N'=128.  256-thread blocks own 64 rows; each wave owns tile pairs (p,q) so
// the sigmoid gate is applied in registers.
// ---------------------------------------------------------------------------
__global__ void __launch_bounds__(256)
stconv_stage3(const unsigned short* __restrict__ h2b,
              const float* __restrict__ W2,
              const float* __restrict__ b2,
              float* __restrict__ out)
{
  __shared__ unsigned short sW2T[128 * W2T_PITCH];  // 18,432 B: W2^T [co][K] bf16
  __shared__ float sB2[128];

  const int tid  = threadIdx.x;
  int blk = blockIdx.x;
  const int nblk = blk & 15;  blk >>= 4;
  const int t = blk % T2;
  const int b = blk / T2;
  const int n0 = nblk * 64;

  for (int i = tid; i < 128 * 64; i += 256) {
    const int co = i >> 6, K = i & 63;
    const float v = (K < 48) ? W2[K * 128 + co] : 0.0f;   // zero-pad K 48..63
    sW2T[co * W2T_PITCH + K] = f2bf(v);
  }
  for (int i = tid; i < 128; i += 256) sB2[i] = b2[i];
  __syncthreads();

  const int lane = tid & 31, lo = lane & 15, hi = lane >> 4;
  const int wid  = tid >> 5;
  const int mt   = wid >> 1;              // 4 row-tiles of 16
  const int c0   = (wid & 1) * 2;         // {0,1,4,5} or {2,3,6,7}
  const int cts[4] = {c0, c0 + 1, c0 + 4, c0 + 5};
  const int row  = n0 + mt * 16 + lo;

  const v8f vz = {0.f,0.f,0.f,0.f,0.f,0.f,0.f,0.f};
  v8f d[4] = {vz, vz, vz, vz};

  const unsigned short* r0 = h2b + ((size_t)(b * T1 + t + 0) * NN + row) * CSPAT + 8 * hi;
  const unsigned short* r1 = h2b + ((size_t)(b * T1 + t + 1) * NN + row) * CSPAT + 8 * hi;
  const unsigned short* r2 = h2b + ((size_t)(b * T1 + t + 2) * NN + row) * CSPAT + 8 * hi;

  // K-tile 0: K 0..31 = [kt=0 ci 0..15 | kt=1 ci 0..15]
  {
    BfTile A; A.u[0] = *(const u32x4*)r0; A.u[1] = *(const u32x4*)r1;
    #pragma unroll
    for (int i = 0; i < 4; ++i) {
      BfTile B;
      const unsigned short* bp = &sW2T[(cts[i] * 16 + lo) * W2T_PITCH + 8 * hi];
      B.u[0] = *(const u32x4*)bp; B.u[1] = *(const u32x4*)(bp + 16);
      d[i] = __builtin_amdgcn_wmma_f32_16x16x32_bf16(
          false, A.bf, false, B.bf, (short)0, d[i], false, false);
    }
  }
  // K-tile 1: K 32..63 = [kt=2 ci 0..15 | zero pad]
  {
    BfTile A; A.u[0] = *(const u32x4*)r2;
    const u32x4 z = {0u, 0u, 0u, 0u};
    A.u[1] = z;
    #pragma unroll
    for (int i = 0; i < 4; ++i) {
      BfTile B;
      const unsigned short* bp = &sW2T[(cts[i] * 16 + lo) * W2T_PITCH + 32 + 8 * hi];
      B.u[0] = *(const u32x4*)bp; B.u[1] = *(const u32x4*)(bp + 16);
      d[i] = __builtin_amdgcn_wmma_f32_16x16x32_bf16(
          false, A.bf, false, B.bf, (short)0, d[i], false, false);
    }
  }

  // GLU epilogue: (d[i], d[i+2]) are the (lhs, rhs) channel tiles
  #pragma unroll
  for (int i = 0; i < 2; ++i) {
    const int coP = cts[i] * 16 + lo;       // 0..63
    const int coQ = coP + 64;
    #pragma unroll
    for (int v = 0; v < 8; ++v) {
      const int r = n0 + mt * 16 + 8 * hi + v;
      const float pv = d[i][v]     + sB2[coP];
      const float qv = d[i + 2][v] + sB2[coQ];
      const float g  = pv * (1.0f / (1.0f + __expf(-qv)));
      out[((size_t)(b * T2 + t) * NN + r) * 64 + coP] = g;
    }
  }
}

// ---------------------------------------------------------------------------
extern "C" void kernel_launch(void* const* d_in, const int* in_sizes, int n_in,
                              void* d_out, int out_size, void* d_ws, size_t ws_size,
                              hipStream_t stream) {
  (void)in_sizes; (void)n_in; (void)out_size; (void)ws_size;
  const float* x     = (const float*)d_in[0];
  const float* cheb  = (const float*)d_in[1];
  const float* W1    = (const float*)d_in[2];
  const float* b1    = (const float*)d_in[3];
  const float* Theta = (const float*)d_in[4];
  const float* W2    = (const float*)d_in[5];
  const float* b2    = (const float*)d_in[6];
  float* out = (float*)d_out;

  // workspace layout: chebb (bf16, 6.3MB) | h2b (bf16, 32.5MB)
  unsigned short* chebb = (unsigned short*)d_ws;
  unsigned short* h2b   =
      (unsigned short*)((char*)d_ws + (size_t)KC * NN * NN * sizeof(unsigned short));

  const int ncheb = KC * NN * NN;
  cheb_to_bf16<<<(ncheb + 255) / 256, 256, 0, stream>>>(cheb, chebb, ncheb);
  stconv_stage12<<<16 * T1, 1024, 0, stream>>>(x, chebb, W1, b1, Theta, h2b);
  stconv_stage3<<<16 * T2 * 16, 256, 0, stream>>>(h2b, W2, b2, out);
}
